// Classifier_65111704207481
// MI455X (gfx1250) — matrix operand order repaired and checked by reading, not compiled, not run on hardware
//
#include <hip/hip_runtime.h>
#include <hip/hip_bf16.h>
#include <math.h>

// ---------------- types ----------------
typedef __attribute__((ext_vector_type(16))) __bf16 v16bf;
typedef __attribute__((ext_vector_type(8)))  __bf16 v8bf;
typedef __attribute__((ext_vector_type(8)))  float  v8f;

#define B_ROWS 8192
#define FEAT   2048
#define BM 128
#define BN 128
#define BK 32
#define LDA 40   // padded LDS row stride (bf16 elems): 20 dwords -> conflict-friendly

// ws layout: xn[8192] f32 | yn[8192] f32 | counters[2] u32 | pad to 256 | 4 bf16 planes
#define WS_PLANES_OFF 65792ull
#define PLANE_ELEMS   ((size_t)B_ROWS * FEAT)

// Pack the bf16 (high-16) parts of two floats into one dword: [f0.hi16 | f1.hi16<<16]
__device__ __forceinline__ unsigned int pack_hi2(float f0, float f1) {
    return __builtin_amdgcn_perm(__float_as_uint(f1), __float_as_uint(f0), 0x07060302u);
}
// Residuals lo = f - bf16_trunc(f) (exact Dekker split), truncated to bf16 and packed.
__device__ __forceinline__ unsigned int pack_lo2(float f0, float f1) {
    float l0 = f0 - __uint_as_float(__float_as_uint(f0) & 0xFFFF0000u);
    float l1 = f1 - __uint_as_float(__float_as_uint(f1) & 0xFFFF0000u);
    return __builtin_amdgcn_perm(__float_as_uint(l1), __float_as_uint(l0), 0x07060302u);
}

// Two async B128 copies global->LDS (32 bytes per lane); offset: applies to both sides.
__device__ __forceinline__ void async_pair(unsigned lds_off, const void* gaddr) {
    asm volatile("global_load_async_to_lds_b128 %0, %1, off\n\t"
                 "global_load_async_to_lds_b128 %0, %1, off offset:16"
                 :: "v"(lds_off), "v"(gaddr) : "memory");
}
__device__ __forceinline__ unsigned lds_addr32(const void* p) {
    return (unsigned)(uintptr_t)p;   // generic LDS pointer: addr[31:0] is the LDS offset
}

// ---------------- kernel 0: one-shot fp32 -> (hi,lo) bf16 plane conversion ----------------
__global__ __launch_bounds__(256) void cls_convert_kernel(
    const float* __restrict__ src,
    unsigned short* __restrict__ hi, unsigned short* __restrict__ lo)
{
    size_t g = (size_t)blockIdx.x * 256 + threadIdx.x;   // uint4-group index
    const float4* s = (const float4*)src + g * 2;
    float4 a = s[0], b = s[1];
    uint4 h = make_uint4(pack_hi2(a.x, a.y), pack_hi2(a.z, a.w),
                         pack_hi2(b.x, b.y), pack_hi2(b.z, b.w));
    uint4 l = make_uint4(pack_lo2(a.x, a.y), pack_lo2(a.z, a.w),
                         pack_lo2(b.x, b.y), pack_lo2(b.z, b.w));
    ((uint4*)hi)[g] = h;
    ((uint4*)lo)[g] = l;
}

// ---------------- kernel 1: row norms (+ counter init) ----------------
__global__ __launch_bounds__(256) void cls_norms_kernel(
    const float* __restrict__ Z, const float* __restrict__ Y,
    float* __restrict__ xn, float* __restrict__ yn, unsigned* __restrict__ counters)
{
    const int row = blockIdx.x;
    const int tid = threadIdx.x;
    if (row == 0 && tid < 2) counters[tid] = 0u;

    const float4* zr = (const float4*)(Z + (size_t)row * FEAT);
    const float4* yr = (const float4*)(Y + (size_t)row * FEAT);
    float sz = 0.f, sy = 0.f;
#pragma unroll
    for (int c = tid; c < FEAT / 4; c += 256) {
        float4 a = zr[c]; sz += a.x * a.x + a.y * a.y + a.z * a.z + a.w * a.w;
        float4 b = yr[c]; sy += b.x * b.x + b.y * b.y + b.z * b.z + b.w * b.w;
    }
#pragma unroll
    for (int off = 16; off > 0; off >>= 1) {
        sz += __shfl_down(sz, off, 32);
        sy += __shfl_down(sy, off, 32);
    }
    __shared__ float redz[8], redy[8];
    if ((tid & 31) == 0) { redz[tid >> 5] = sz; redy[tid >> 5] = sy; }
    __syncthreads();
    if (tid == 0) {
        float tz = 0.f, ty = 0.f;
#pragma unroll
        for (int w = 0; w < 8; ++w) { tz += redz[w]; ty += redy[w]; }
        xn[row] = sqrtf(tz);
        yn[row] = sqrtf(ty);
    }
}

// ---- shared epilogue ----
__device__ __forceinline__ void gemm_epilogue(
    v8f acc[4][2], int m0, int n0, int wm, int wn, int lane,
    const float* __restrict__ yn, const float* __restrict__ xn, float* __restrict__ sim)
{
    const int lane15 = lane & 15;
    const int mrow_off = (lane >> 4) * 8;  // C/D layout: lanes 16-31 hold M+8
#pragma unroll
    for (int tm = 0; tm < 4; ++tm) {
        int gm0 = m0 + wm * 64 + tm * 16;
#pragma unroll
        for (int tn = 0; tn < 2; ++tn) {
            int gn = n0 + wn * 32 + tn * 16 + lane15;
            float xnj = xn[gn];
#pragma unroll
            for (int r = 0; r < 8; ++r) {
                int gm = gm0 + mrow_off + r;
                float denom = fmaxf(yn[gm] * xnj, 1e-8f);
                sim[(size_t)gm * B_ROWS + gn] = acc[tm][tn][r] / denom;
            }
        }
    }
}

// ---- shared fragment loads + 24 WMMAs for one K-step ----
__device__ __forceinline__ void gemm_step(
    const unsigned short* __restrict__ AhiP, const unsigned short* __restrict__ AloP,
    const unsigned short* __restrict__ BhiP, const unsigned short* __restrict__ BloP,
    int wm, int wn, int lane, v8f acc[4][2])
{
    const int lane15 = lane & 15;
    const int koffA  = (lane >> 4) * 8;   // A frag: K 0-7/16-23 or 8-15/24-31
    const int koffB  = (lane >> 4) * 8;   // B frag: K 0-7/16-23 or 8-15/24-31

    v16bf ahiF[4], aloF[4], bhiF[2], bloF[2];
#pragma unroll
    for (int tm = 0; tm < 4; ++tm) {
        int rb = (wm * 64 + tm * 16 + lane15) * LDA;
        v8bf h0 = *(const v8bf*)&AhiP[rb + koffA];
        v8bf h1 = *(const v8bf*)&AhiP[rb + koffA + 16];
        ahiF[tm] = __builtin_shufflevector(h0, h1, 0,1,2,3,4,5,6,7,8,9,10,11,12,13,14,15);
        v8bf l0 = *(const v8bf*)&AloP[rb + koffA];
        v8bf l1 = *(const v8bf*)&AloP[rb + koffA + 16];
        aloF[tm] = __builtin_shufflevector(l0, l1, 0,1,2,3,4,5,6,7,8,9,10,11,12,13,14,15);
    }
#pragma unroll
    for (int tn = 0; tn < 2; ++tn) {
        int cb = (wn * 32 + tn * 16 + lane15) * LDA;
        v8bf h0 = *(const v8bf*)&BhiP[cb + koffB];
        v8bf h1 = *(const v8bf*)&BhiP[cb + koffB + 16];
        bhiF[tn] = __builtin_shufflevector(h0, h1, 0,1,2,3,4,5,6,7,8,9,10,11,12,13,14,15);
        v8bf l0 = *(const v8bf*)&BloP[cb + koffB];
        v8bf l1 = *(const v8bf*)&BloP[cb + koffB + 16];
        bloF[tn] = __builtin_shufflevector(l0, l1, 0,1,2,3,4,5,6,7,8,9,10,11,12,13,14,15);
    }
#pragma unroll
    for (int tm = 0; tm < 4; ++tm) {
#pragma unroll
        for (int tn = 0; tn < 2; ++tn) {
            acc[tm][tn] = __builtin_amdgcn_wmma_f32_16x16x32_bf16(
                false, ahiF[tm], false, bhiF[tn], (short)0, acc[tm][tn], false, false);
            acc[tm][tn] = __builtin_amdgcn_wmma_f32_16x16x32_bf16(
                false, ahiF[tm], false, bloF[tn], (short)0, acc[tm][tn], false, false);
            acc[tm][tn] = __builtin_amdgcn_wmma_f32_16x16x32_bf16(
                false, aloF[tm], false, bhiF[tn], (short)0, acc[tm][tn], false, false);
        }
    }
}

// ---------------- kernel 2a: pre-converted GEMM, async-to-LDS + double buffering ----------------
__global__ __launch_bounds__(256) void cls_gemm_pre_kernel(
    const unsigned short* __restrict__ Yhi, const unsigned short* __restrict__ Ylo,
    const unsigned short* __restrict__ Zhi, const unsigned short* __restrict__ Zlo,
    const float* __restrict__ yn, const float* __restrict__ xn,
    float* __restrict__ sim)
{
    __shared__ __attribute__((aligned(16))) unsigned short Ah[2][BM * LDA];
    __shared__ __attribute__((aligned(16))) unsigned short Al[2][BM * LDA];
    __shared__ __attribute__((aligned(16))) unsigned short Bh[2][BN * LDA];
    __shared__ __attribute__((aligned(16))) unsigned short Bl[2][BN * LDA];

    const int tid  = threadIdx.x;
    const int lane = tid & 31;
    const int wave = tid >> 5;
    const int wm   = wave & 1;
    const int wn   = wave >> 1;
    const int m0   = blockIdx.y * BM;
    const int n0   = blockIdx.x * BN;

    const int srow  = tid >> 1;         // staged row
    const int shalf = (tid & 1) * 16;   // staged K half (elems)

    // per-thread async destinations (LDS byte addresses), per buffer
    unsigned ldsAh[2], ldsAl[2], ldsBh[2], ldsBl[2];
#pragma unroll
    for (int p = 0; p < 2; ++p) {
        ldsAh[p] = lds_addr32(&Ah[p][srow * LDA + shalf]);
        ldsAl[p] = lds_addr32(&Al[p][srow * LDA + shalf]);
        ldsBh[p] = lds_addr32(&Bh[p][srow * LDA + shalf]);
        ldsBl[p] = lds_addr32(&Bl[p][srow * LDA + shalf]);
    }
    // per-thread global sources (advance by kb elements)
    const unsigned short* gAh = Yhi + (size_t)(m0 + srow) * FEAT + shalf;
    const unsigned short* gAl = Ylo + (size_t)(m0 + srow) * FEAT + shalf;
    const unsigned short* gBh = Zhi + (size_t)(n0 + srow) * FEAT + shalf;
    const unsigned short* gBl = Zlo + (size_t)(n0 + srow) * FEAT + shalf;

    v8f acc[4][2];
#pragma unroll
    for (int tm = 0; tm < 4; ++tm)
#pragma unroll
        for (int tn = 0; tn < 2; ++tn)
            acc[tm][tn] = (v8f){};

    // prefetch tile 0 into buffer 0
    async_pair(ldsAh[0], gAh);
    async_pair(ldsAl[0], gAl);
    async_pair(ldsBh[0], gBh);
    async_pair(ldsBl[0], gBl);

    int p = 0;
    for (int kb = 0; kb < FEAT; kb += BK) {
        // my async writes into buf p and my previous frag reads are complete
        asm volatile("s_wait_asynccnt 0x0" ::: "memory");
        asm volatile("s_wait_dscnt 0x0" ::: "memory");
        __syncthreads();   // now buf p is globally filled, buf p^1 globally free

        if (kb + BK < FEAT) {  // prefetch next tile into the other buffer (overlaps WMMAs)
            async_pair(ldsAh[p ^ 1], gAh + kb + BK);
            async_pair(ldsAl[p ^ 1], gAl + kb + BK);
            async_pair(ldsBh[p ^ 1], gBh + kb + BK);
            async_pair(ldsBl[p ^ 1], gBl + kb + BK);
        }

        gemm_step(Ah[p], Al[p], Bh[p], Bl[p], wm, wn, lane, acc);
        p ^= 1;
    }

    gemm_epilogue(acc, m0, n0, wm, wn, lane, yn, xn, sim);
}

// ---------------- kernel 2b: fallback GEMM with inline fp32->bf16 split ----------------
__global__ __launch_bounds__(256) void cls_gemm_fallback_kernel(
    const float* __restrict__ Yg, const float* __restrict__ Zg,
    const float* __restrict__ yn, const float* __restrict__ xn,
    float* __restrict__ sim)
{
    __shared__ __attribute__((aligned(16))) unsigned short Ahi[BM * LDA];
    __shared__ __attribute__((aligned(16))) unsigned short Alo[BM * LDA];
    __shared__ __attribute__((aligned(16))) unsigned short Bhi[BN * LDA];
    __shared__ __attribute__((aligned(16))) unsigned short Blo[BN * LDA];

    const int tid  = threadIdx.x;
    const int lane = tid & 31;
    const int wave = tid >> 5;
    const int wm   = wave & 1;
    const int wn   = wave >> 1;
    const int m0   = blockIdx.y * BM;
    const int n0   = blockIdx.x * BN;

    const int lrow  = tid >> 1;
    const int lhalf = (tid & 1) * 16;

    uint4* ahp = (uint4*)&Ahi[lrow * LDA + lhalf];
    uint4* alp = (uint4*)&Alo[lrow * LDA + lhalf];
    uint4* bhp = (uint4*)&Bhi[lrow * LDA + lhalf];
    uint4* blp = (uint4*)&Blo[lrow * LDA + lhalf];

    v8f acc[4][2];
#pragma unroll
    for (int tm = 0; tm < 4; ++tm)
#pragma unroll
        for (int tn = 0; tn < 2; ++tn)
            acc[tm][tn] = (v8f){};

    for (int kb = 0; kb < FEAT; kb += BK) {
        const float4* yp = (const float4*)(Yg + (size_t)(m0 + lrow) * FEAT + kb + lhalf);
        const float4* zp = (const float4*)(Zg + (size_t)(n0 + lrow) * FEAT + kb + lhalf);
        float4 yq0 = yp[0], yq1 = yp[1], yq2 = yp[2], yq3 = yp[3];
        float4 zq0 = zp[0], zq1 = zp[1], zq2 = zp[2], zq3 = zp[3];

        uint4 yh0 = make_uint4(pack_hi2(yq0.x, yq0.y), pack_hi2(yq0.z, yq0.w),
                               pack_hi2(yq1.x, yq1.y), pack_hi2(yq1.z, yq1.w));
        uint4 yh1 = make_uint4(pack_hi2(yq2.x, yq2.y), pack_hi2(yq2.z, yq2.w),
                               pack_hi2(yq3.x, yq3.y), pack_hi2(yq3.z, yq3.w));
        uint4 yl0 = make_uint4(pack_lo2(yq0.x, yq0.y), pack_lo2(yq0.z, yq0.w),
                               pack_lo2(yq1.x, yq1.y), pack_lo2(yq1.z, yq1.w));
        uint4 yl1 = make_uint4(pack_lo2(yq2.x, yq2.y), pack_lo2(yq2.z, yq2.w),
                               pack_lo2(yq3.x, yq3.y), pack_lo2(yq3.z, yq3.w));
        uint4 zh0 = make_uint4(pack_hi2(zq0.x, zq0.y), pack_hi2(zq0.z, zq0.w),
                               pack_hi2(zq1.x, zq1.y), pack_hi2(zq1.z, zq1.w));
        uint4 zh1 = make_uint4(pack_hi2(zq2.x, zq2.y), pack_hi2(zq2.z, zq2.w),
                               pack_hi2(zq3.x, zq3.y), pack_hi2(zq3.z, zq3.w));
        uint4 zl0 = make_uint4(pack_lo2(zq0.x, zq0.y), pack_lo2(zq0.z, zq0.w),
                               pack_lo2(zq1.x, zq1.y), pack_lo2(zq1.z, zq1.w));
        uint4 zl1 = make_uint4(pack_lo2(zq2.x, zq2.y), pack_lo2(zq2.z, zq2.w),
                               pack_lo2(zq3.x, zq3.y), pack_lo2(zq3.z, zq3.w));

        __syncthreads();
        ahp[0] = yh0; ahp[1] = yh1;
        alp[0] = yl0; alp[1] = yl1;
        bhp[0] = zh0; bhp[1] = zh1;
        blp[0] = zl0; blp[1] = zl1;
        __syncthreads();

        gemm_step(Ahi, Alo, Bhi, Blo, wm, wn, lane, acc);
    }

    gemm_epilogue(acc, m0, n0, wm, wn, lane, yn, xn, sim);
}

// ---------------- kernel 3: per-row rank of the diagonal ----------------
__global__ __launch_bounds__(256) void cls_rowstats_kernel(
    const float* __restrict__ sim, unsigned* __restrict__ counters)
{
    const int i   = blockIdx.x;
    const int tid = threadIdx.x;
    const float* row = sim + (size_t)i * B_ROWS;
    const float diag = row[i];
    int cnt = 0;
#pragma unroll 4
    for (int j = tid; j < B_ROWS; j += 256) cnt += (row[j] > diag) ? 1 : 0;
#pragma unroll
    for (int off = 16; off > 0; off >>= 1) cnt += __shfl_down(cnt, off, 32);
    __shared__ int red[8];
    if ((tid & 31) == 0) red[tid >> 5] = cnt;
    __syncthreads();
    if (tid == 0) {
        int t = 0;
#pragma unroll
        for (int w = 0; w < 8; ++w) t += red[w];
        if (t == 0) atomicAdd(&counters[0], 1u);   // strict max at diagonal -> argmax == i
        if (t < 10) atomicAdd(&counters[1], 1u);   // fewer than 10 greater -> in top-10
    }
}

// ---------------- kernel 4: finalize scalars ----------------
__global__ void cls_finalize_kernel(const unsigned* __restrict__ counters, float* __restrict__ out)
{
    out[0] = (float)counters[0] * (1.0f / (float)B_ROWS);
    out[1] = (float)counters[1] * (1.0f / (float)B_ROWS);
}

extern "C" void kernel_launch(void* const* d_in, const int* in_sizes, int n_in,
                              void* d_out, int out_size, void* d_ws, size_t ws_size,
                              hipStream_t stream) {
    const float* Z = (const float*)d_in[0];  // x rows (columns j of sim)
    const float* Y = (const float*)d_in[1];  // y rows (rows i of sim)
    float* out = (float*)d_out;
    float* sim = out + 2;                    // (top1, top10, sim) flattened

    float*    xn       = (float*)d_ws;
    float*    yn       = xn + B_ROWS;
    unsigned* counters = (unsigned*)(yn + B_ROWS);

    unsigned short* Yhi = (unsigned short*)((char*)d_ws + WS_PLANES_OFF);
    unsigned short* Ylo = Yhi + PLANE_ELEMS;
    unsigned short* Zhi = Ylo + PLANE_ELEMS;
    unsigned short* Zlo = Zhi + PLANE_ELEMS;
    const size_t ws_needed = WS_PLANES_OFF + 4 * PLANE_ELEMS * sizeof(unsigned short);

    cls_norms_kernel<<<B_ROWS, 256, 0, stream>>>(Z, Y, xn, yn, counters);

    dim3 grid(B_ROWS / BN, B_ROWS / BM);
    const int convBlocks = (int)(PLANE_ELEMS / 8 / 256);   // 8 floats per thread

    if (ws_size >= ws_needed) {
        cls_convert_kernel<<<convBlocks, 256, 0, stream>>>(Y, Yhi, Ylo);
        cls_convert_kernel<<<convBlocks, 256, 0, stream>>>(Z, Zhi, Zlo);
        cls_gemm_pre_kernel<<<grid, 256, 0, stream>>>(Yhi, Ylo, Zhi, Zlo, yn, xn, sim);
    } else {
        cls_gemm_fallback_kernel<<<grid, 256, 0, stream>>>(Y, Z, yn, xn, sim);
    }

    cls_rowstats_kernel<<<B_ROWS, 256, 0, stream>>>(sim, counters);
    cls_finalize_kernel<<<1, 1, 0, stream>>>(counters, out);
}
